// differentiableSuperpixelTokenizer_28132035788970
// MI455X (gfx1250) — matrix operand order repaired and checked by reading, not compiled, not run on hardware
//
#include <hip/hip_runtime.h>
#include <hip/hip_bf16.h>

typedef __attribute__((ext_vector_type(2)))  float  v2f;
typedef __attribute__((ext_vector_type(8)))  float  v8f;
typedef __attribute__((ext_vector_type(8)))  __bf16 v8bf;
typedef __attribute__((ext_vector_type(16))) __bf16 v16bf;

#define Bn 4
#define Cn 3
#define Hn 224
#define Wn 224
#define Sn 196
#define En 768
#define NHn 4
#define Dn 192
#define Pn 256
#define GRIDn 14
#define PATCHn 16
#define SSTR 257   // LDS row stride (odd) -> conflict-free row & column passes

// ---------------------------------------------------------------------------
// Kernel 1: fold feature projection into the QKV projection.
// Wc[r][a] (r in [0,2304), a in [0,4)):
//   a<3 : sum_e in_proj_w[r][e] * w_fp[e][a]
//   a==3: sum_e in_proj_w[r][e] * b_fp[e] + in_proj_b[r]    (augmented bias col)
// ---------------------------------------------------------------------------
__global__ void k_combine(const float* __restrict__ w_fp, const float* __restrict__ b_fp,
                          const float* __restrict__ ipw,  const float* __restrict__ ipb,
                          float* __restrict__ Wc) {
  int t = blockIdx.x * blockDim.x + threadIdx.x;
  if (t >= 3 * En * 4) return;
  int r = t >> 2, a = t & 3;
  const float* wr = ipw + (size_t)r * En;
  float acc = 0.f;
  if (a < 3) {
    for (int e = 0; e < En; ++e) acc += wr[e] * w_fp[e * 3 + a];
  } else {
    for (int e = 0; e < En; ++e) acc += wr[e] * b_fp[e];
    acc += ipb[r];
  }
  Wc[r * 4 + a] = acc;
}

// ---------------------------------------------------------------------------
// Kernel 2: per-head augmented 4x4 score matrix  M~h = Qh_aug^T * Kh_aug,
// scaled by 1/sqrt(D).  Mt[h*16 + a*4 + b].
// ---------------------------------------------------------------------------
__global__ void k_mtilde(const float* __restrict__ Wc, float* __restrict__ Mt) {
  int t = threadIdx.x;
  if (t >= 64) return;
  int h = t >> 4, a = (t >> 2) & 3, bcol = t & 3;
  const float* q = Wc + (size_t)(h * Dn) * 4;
  const float* k = Wc + (size_t)(En + h * Dn) * 4;
  float acc = 0.f;
  for (int d = 0; d < Dn; ++d) acc += q[d * 4 + a] * k[d * 4 + bcol];
  Mt[t] = acc * rsqrtf((float)Dn);
}

// ---------------------------------------------------------------------------
// Kernel 3: f32 -> bf16 conversion (out_proj weight)
// ---------------------------------------------------------------------------
__global__ void k_cvt(const float* __restrict__ src, __hip_bfloat16* __restrict__ dst, int n) {
  int i = blockIdx.x * blockDim.x + threadIdx.x;
  if (i < n) dst[i] = __float2bfloat16(src[i]);
}

// ---------------------------------------------------------------------------
// Kernel 4: per-(b,s) attention, fully collapsed through the rank-4 structure.
//   scores = Gaug * M~h * Gaug^T   via V_WMMA_F32_16X16X4_F32 (1 WMMA / tile)
//   softmax rows, column-sums, tvec = Gaug^T * c / P, omean_h = V~h * tvec
// 256 threads (8 wave32), one workgroup per (b,s).
// ---------------------------------------------------------------------------
__global__ __launch_bounds__(256, 1)
void k_attn(const float* __restrict__ img, const float* __restrict__ Wc,
            const float* __restrict__ Mt, __hip_bfloat16* __restrict__ omean) {
  __shared__ float sc[Pn * SSTR];     // 263168 B scores
  __shared__ float gaug[Pn][4];       // augmented pixels (r,g,b,1)
  __shared__ float gm[Pn][4];         // Gaug @ M~h
  __shared__ float rrz[Pn];           // 1/Z per row
  __shared__ float csum[Pn];          // column sums of A
  __shared__ float msh[64];
  __shared__ float tvec[4];

  const int tid = threadIdx.x;
  const int bs  = blockIdx.x;
  const int b = bs / Sn, s = bs % Sn;
  // stable-argsort of the fixed grid == raster order inside each 16x16 patch
  const int row = (s / GRIDn) * PATCHn + (tid >> 4);
  const int col = (s % GRIDn) * PATCHn + (tid & 15);

  const float* px = img + (size_t)b * Cn * Hn * Wn + (size_t)row * Wn + col;
  gaug[tid][0] = px[0];
  gaug[tid][1] = px[Hn * Wn];
  gaug[tid][2] = px[2 * Hn * Wn];
  gaug[tid][3] = 1.0f;
  if (tid < 64) msh[tid] = Mt[tid];
  __syncthreads();

  const int wv = tid >> 5, lane = tid & 31;
  const int l15  = lane & 15;
  const int kb   = (lane < 16) ? 0 : 2;  // f32 A/B frag: lane half selects K pair
  const int rsel = (lane < 16) ? 0 : 8;  // f32 C/D frag: lane half selects row+8

  for (int h = 0; h < NHn; ++h) {
    { // GM = Gaug @ M~h  (each thread: its own row)
      float g0 = gaug[tid][0], g1 = gaug[tid][1], g2 = gaug[tid][2], g3 = gaug[tid][3];
      const float* m = msh + h * 16;
#pragma unroll
      for (int j = 0; j < 4; ++j)
        gm[tid][j] = g0 * m[j] + g1 * m[4 + j] + g2 * m[8 + j] + g3 * m[12 + j];
    }
    __syncthreads();

    // 256 16x16 score tiles, 8 waves * 32 tiles each.
    // Scalar (compile-time) trip count: no exec-mask loop around the WMMAs;
    // unroll x4 so the scheduler can pipeline WMMAs against LDS traffic.
#pragma unroll 4
    for (int i = 0; i < 32; ++i) {
      const int t  = (i << 3) + wv;   // uniform per wave
      const int ti = t >> 4, tj = t & 15;
      const int ar = ti * 16 + l15;   // A row (M)
      const int bc = tj * 16 + l15;   // B col (N)
      v2f A;  A.x  = gm[ar][kb];   A.y  = gm[ar][kb + 1];
      v2f Bm; Bm.x = gaug[bc][kb]; Bm.y = gaug[bc][kb + 1];
      v8f Cz = {0.f, 0.f, 0.f, 0.f, 0.f, 0.f, 0.f, 0.f};
      v8f Dv = __builtin_amdgcn_wmma_f32_16x16x4_f32(false, A, false, Bm,
                                                     (short)0, Cz, false, false);
      const int orow = ti * 16 + rsel;
#pragma unroll
      for (int v = 0; v < 8; ++v)
        sc[(orow + v) * SSTR + bc] = Dv[v];
    }
    __syncthreads();

    { // row softmax (thread = row): store exp in place, keep 1/Z
      float* rp = sc + tid * SSTR;
      float mx = -3.4e38f;
      for (int j = 0; j < Pn; ++j) mx = fmaxf(mx, rp[j]);
      float z = 0.f;
      for (int j = 0; j < Pn; ++j) { float e = __expf(rp[j] - mx); rp[j] = e; z += e; }
      rrz[tid] = 1.0f / z;
    }
    __syncthreads();

    { // column sums of normalized A (thread = col)
      float cs = 0.f;
      for (int i = 0; i < Pn; ++i) cs += sc[i * SSTR + tid] * rrz[i];
      csum[tid] = cs;
    }
    __syncthreads();

    if (tid < 4) { // tvec = Gaug^T * c / P   (tvec[3] == 1 -> v-bias applied once)
      float acc = 0.f;
      for (int j = 0; j < Pn; ++j) acc += csum[j] * gaug[j][tid];
      tvec[tid] = acc * (1.0f / Pn);
    }
    __syncthreads();

    if (tid < Dn) { // omean slice for this head: V~h @ tvec
      const float* vr = Wc + (size_t)(2 * En + h * Dn + tid) * 4;
      float o = vr[0] * tvec[0] + vr[1] * tvec[1] + vr[2] * tvec[2] + vr[3] * tvec[3];
      omean[(size_t)bs * En + h * Dn + tid] = __float2bfloat16(o);
    }
    __syncthreads();
  }
}

// ---------------------------------------------------------------------------
// Kernel 5: out = omean[784,768] @ out_proj_w^T + b  via bf16 WMMA 16x16x32.
// One wave per 16x16 output tile; grid (48 N-tiles, 49 M-tiles). 784 = 49*16.
// ---------------------------------------------------------------------------
__global__ __launch_bounds__(32)
void k_outproj(const __bf16* __restrict__ Abf, const __bf16* __restrict__ Bbf,
               const float* __restrict__ bias, float* __restrict__ out) {
  const int tn = blockIdx.x;          // N tile
  const int tm = blockIdx.y;          // M tile
  const int lane = threadIdx.x;
  const int l15 = lane & 15;
  const int m = tm * 16 + l15;
  const int n = tn * 16 + l15;
  const int alo = (lane < 16) ? 0 : 16;  // A frag: lane half -> K block 0/16
  const int bko = (lane < 16) ? 0 : 16;  // B frag: lane half -> K block 0/16

  v8f acc = {0.f, 0.f, 0.f, 0.f, 0.f, 0.f, 0.f, 0.f};
#pragma unroll 4
  for (int k0 = 0; k0 < En; k0 += 32) {
    const __bf16* ap = Abf + (size_t)m * En + k0 + alo;
    v8bf a0 = *(const v8bf*)(ap);        // K = lo .. lo+7
    v8bf a1 = *(const v8bf*)(ap + 8);    // K = lo+8 .. lo+15
    v16bf Af = __builtin_shufflevector(a0, a1, 0, 1, 2, 3, 4, 5, 6, 7,
                                       8, 9, 10, 11, 12, 13, 14, 15);
    // B[k][n] = Wout[n][k]: contiguous 16 bf16 per lane (32B aligned)
    v16bf Bf = *(const v16bf*)(Bbf + (size_t)n * En + k0 + bko);
    acc = __builtin_amdgcn_wmma_f32_16x16x32_bf16(false, Af, false, Bf,
                                                  (short)0, acc, false, false);
  }
  const int rbase = tm * 16 + ((lane < 16) ? 0 : 8);
  const float bv = bias[n];
#pragma unroll
  for (int v = 0; v < 8; ++v)
    out[(size_t)(rbase + v) * En + n] = acc[v] + bv;
}

// ---------------------------------------------------------------------------
extern "C" void kernel_launch(void* const* d_in, const int* in_sizes, int n_in,
                              void* d_out, int out_size, void* d_ws, size_t ws_size,
                              hipStream_t stream) {
  const float* img  = (const float*)d_in[0];
  // d_in[1] (segments) is the fixed 14x14 grid; addressed analytically.
  const float* w_fp = (const float*)d_in[2];
  const float* b_fp = (const float*)d_in[3];
  const float* ipw  = (const float*)d_in[4];
  const float* ipb  = (const float*)d_in[5];
  const float* opw  = (const float*)d_in[6];
  const float* opb  = (const float*)d_in[7];

  char* ws = (char*)d_ws;
  float* Wc = (float*)ws;                                    // 2304*4 f32 = 36864 B
  float* Mt = (float*)(ws + 2304 * 4 * sizeof(float));       // 64 f32    = 256 B
  __hip_bfloat16* omeanBf = (__hip_bfloat16*)(ws + 36864 + 256);        // 784*768 bf16
  __hip_bfloat16* WoutBf  = omeanBf + (size_t)784 * 768;                // 768*768 bf16

  k_combine<<<(3 * En * 4 + 255) / 256, 256, 0, stream>>>(w_fp, b_fp, ipw, ipb, Wc);
  k_mtilde<<<1, 64, 0, stream>>>(Wc, Mt);
  k_cvt<<<(En * En + 255) / 256, 256, 0, stream>>>(opw, WoutBf, En * En);
  k_attn<<<Bn * Sn, 256, 0, stream>>>(img, Wc, Mt, omeanBf);
  k_outproj<<<dim3(En / 16, (Bn * Sn) / 16), 32, 0, stream>>>(
      (const __bf16*)omeanBf, (const __bf16*)WoutBf, opb, (float*)d_out);
}
